// GNN_51376398794850
// MI455X (gfx1250) — compile-verified, matrix operand order checked by reading, not run on hardware
//
#include <hip/hip_runtime.h>
#include <hip/hip_bf16.h>
#include <math.h>

#define N_NODES 50000
#define E_EDGES 800000
#define M_PATHS 3
#define IN_F    256
#define H_HEADS 8
#define D_HEAD  32
#define HD      256
#define SEM     128
#define OUT_F   32

typedef __attribute__((ext_vector_type(16))) __bf16 v16bf;
typedef __attribute__((ext_vector_type(8)))  __bf16 v8bf;
typedef __attribute__((ext_vector_type(8)))  float  v8f;

// ---------------------------------------------------------------- helpers
__device__ __forceinline__ __bf16 f2bf(float f) {
    unsigned u = __float_as_uint(f);
    u += 0x7FFFu + ((u >> 16) & 1u);           // round-to-nearest-even
    unsigned short h = (unsigned short)(u >> 16);
    __bf16 r;
    __builtin_memcpy(&r, &h, 2);
    return r;
}

__device__ __forceinline__ void atomicMaxFloat(float* addr, float value) {
    // IEEE-order-preserving int trick; works with -inf init.
    if (__float_as_int(value) >= 0)
        atomicMax((int*)addr, __float_as_int(value));
    else
        atomicMin((unsigned int*)addr, (unsigned int)__float_as_int(value));
}

__device__ __forceinline__ v16bf load_frag(const __bf16* row, int kb, int koff) {
    v8bf lo = *(const v8bf*)(row + kb + koff);
    v8bf hi = *(const v8bf*)(row + kb + koff + 16);
    v16bf f{};
    #pragma unroll
    for (int i = 0; i < 8; ++i) { f[i] = lo[i]; f[i + 8] = hi[i]; }
    return f;
}

// ---------------------------------------------------------------- converts / init
__global__ void cvt_bf16_kernel(const float* __restrict__ src, __bf16* __restrict__ dst, size_t n) {
    size_t i = (size_t)blockIdx.x * blockDim.x + threadIdx.x;
    if (i < n) dst[i] = f2bf(src[i]);
}

// src batch z: [R,C] row-major -> dst batch z: [C,R] row-major (transpose + cast)
__global__ void cvt_bf16_T_kernel(const float* __restrict__ src, __bf16* __restrict__ dst, int R, int C) {
    size_t base = (size_t)blockIdx.z * R * C;
    int i = blockIdx.x * blockDim.x + threadIdx.x;
    if (i < R * C) {
        int r = i / C, c = i % C;
        dst[base + (size_t)c * R + r] = f2bf(src[base + i]);
    }
}

__global__ void fill_u32_kernel(unsigned* __restrict__ p, unsigned v, size_t n) {
    size_t i = (size_t)blockIdx.x * blockDim.x + threadIdx.x;
    if (i < n) p[i] = v;
}

// ---------------------------------------------------------------- bf16 WMMA GEMM
// C[z] = A @ Bt[z]^T.  A: [R,K] bf16 row-major (shared across z),
// Bt: [NC,K] bf16 row-major (pre-transposed B), C: [R,NC] f32.
// One wave computes a 32x32 macro tile = 2x2 v_wmma_f32_16x16x32_bf16 tiles:
// each A/B fragment is reused twice -> 2 b128 loads per WMMA, and the four
// WMMAs per k-step are independent (no D->A/B hazard NOPs).
// R must be a multiple of 16 (remainder-of-32 handled); NC a multiple of 32.
__global__ __launch_bounds__(256)
void gemm_bf16_wmma_kernel(const __bf16* __restrict__ A, const __bf16* __restrict__ Bt,
                           float* __restrict__ C, int R, int K, int NC,
                           long long sBt, long long sC) {
    const int wave = threadIdx.x >> 5;
    const int lane = threadIdx.x & 31;
    const int tM = blockIdx.x * 8 + wave;          // 32-row macro tile index
    if (tM * 32 >= R) return;                      // wave-uniform
    const bool hasRow1 = (tM * 32 + 16) < R;       // wave-uniform (R % 16 == 0)

    const __bf16* Btz = Bt + (size_t)blockIdx.z * sBt;
    float*        Cz  = C  + (size_t)blockIdx.z * sC;

    const int hl   = lane & 15;
    const int koff = (lane >= 16) ? 8 : 0;         // ISA 16-bit A/B lane K split
    const __bf16* Arow0 = A + (size_t)(tM * 32 + hl) * K;
    const __bf16* Arow1 = hasRow1 ? (A + (size_t)(tM * 32 + 16 + hl) * K) : Arow0;
    const __bf16* Brow0 = Btz + (size_t)(blockIdx.y * 32 + hl) * K;
    const __bf16* Brow1 = Btz + (size_t)(blockIdx.y * 32 + 16 + hl) * K;

    v8f acc00 = {}, acc01 = {}, acc10 = {}, acc11 = {};
    for (int kb = 0; kb < K; kb += 32) {
        if (kb + 128 < K) {                        // WGP-scope global_prefetch_b8
            __builtin_prefetch(Arow0 + kb + 128, 0, 3);
            __builtin_prefetch(Arow1 + kb + 128, 0, 3);
            __builtin_prefetch(Brow0 + kb + 128, 0, 3);
            __builtin_prefetch(Brow1 + kb + 128, 0, 3);
        }
        v16bf a0 = load_frag(Arow0, kb, koff);
        v16bf a1 = load_frag(Arow1, kb, koff);
        v16bf b0 = load_frag(Brow0, kb, koff);
        v16bf b1 = load_frag(Brow1, kb, koff);
        acc00 = __builtin_amdgcn_wmma_f32_16x16x32_bf16(false, a0, false, b0, (short)0, acc00, false, false);
        acc01 = __builtin_amdgcn_wmma_f32_16x16x32_bf16(false, a0, false, b1, (short)0, acc01, false, false);
        acc10 = __builtin_amdgcn_wmma_f32_16x16x32_bf16(false, a1, false, b0, (short)0, acc10, false, false);
        acc11 = __builtin_amdgcn_wmma_f32_16x16x32_bf16(false, a1, false, b1, (short)0, acc11, false, false);
    }

    // C layout: VGPR i -> row M = i (lanes 0-15) / i+8 (lanes 16-31), col N = lane&15
    const int rbase = tM * 32 + ((lane >= 16) ? 8 : 0);
    const int col0  = blockIdx.y * 32 + hl;
    #pragma unroll
    for (int i = 0; i < 8; ++i) {
        Cz[(size_t)(rbase + i) * NC + col0]      = acc00[i];
        Cz[(size_t)(rbase + i) * NC + col0 + 16] = acc01[i];
    }
    if (hasRow1) {
        #pragma unroll
        for (int i = 0; i < 8; ++i) {
            Cz[(size_t)(rbase + 16 + i) * NC + col0]      = acc10[i];
            Cz[(size_t)(rbase + 16 + i) * NC + col0 + 16] = acc11[i];
        }
    }
}

// ---------------------------------------------------------------- attention logits
__global__ void el_er_kernel(const float* __restrict__ feat,
                             const float* __restrict__ attn_l, const float* __restrict__ attn_r,
                             float* __restrict__ el, float* __restrict__ er) {
    int idx = blockIdx.x * blockDim.x + threadIdx.x;
    const int total = M_PATHS * N_NODES * H_HEADS;
    if (idx >= total) return;
    int hh = idx % H_HEADS;
    int n  = (idx / H_HEADS) % N_NODES;
    int m  = idx / (H_HEADS * N_NODES);
    const float* f  = feat + ((size_t)m * N_NODES + n) * HD + hh * D_HEAD;
    const float* al = attn_l + (size_t)(m * H_HEADS + hh) * D_HEAD;
    const float* ar = attn_r + (size_t)(m * H_HEADS + hh) * D_HEAD;
    float sl = 0.f, sr = 0.f;
    #pragma unroll 8
    for (int d = 0; d < D_HEAD; ++d) { sl += f[d] * al[d]; sr += f[d] * ar[d]; }
    el[idx] = sl;
    er[idx] = sr;
}

__global__ void edge_max_kernel(const int* __restrict__ ei, const float* __restrict__ el,
                                const float* __restrict__ er, float* __restrict__ emax) {
    int idx = blockIdx.x * blockDim.x + threadIdx.x;
    if (idx >= E_EDGES * H_HEADS) return;
    int m  = blockIdx.z;
    int hh = idx % H_HEADS;
    int e  = idx / H_HEADS;
    int src = ei[(size_t)m * 2 * E_EDGES + e];
    int dst = ei[(size_t)m * 2 * E_EDGES + E_EDGES + e];
    float v = el[((size_t)m * N_NODES + src) * H_HEADS + hh]
            + er[((size_t)m * N_NODES + dst) * H_HEADS + hh];
    v = v > 0.f ? v : 0.2f * v;                    // leaky_relu
    atomicMaxFloat(&emax[((size_t)m * N_NODES + dst) * H_HEADS + hh], v);
}

__global__ void edge_sum_kernel(const int* __restrict__ ei, const float* __restrict__ el,
                                const float* __restrict__ er, const float* __restrict__ emax,
                                float* __restrict__ esum) {
    int idx = blockIdx.x * blockDim.x + threadIdx.x;
    if (idx >= E_EDGES * H_HEADS) return;
    int m  = blockIdx.z;
    int hh = idx % H_HEADS;
    int e  = idx / H_HEADS;
    int src = ei[(size_t)m * 2 * E_EDGES + e];
    int dst = ei[(size_t)m * 2 * E_EDGES + E_EDGES + e];
    size_t dIdx = ((size_t)m * N_NODES + dst) * H_HEADS + hh;
    float v = el[((size_t)m * N_NODES + src) * H_HEADS + hh] + er[dIdx];
    v = v > 0.f ? v : 0.2f * v;
    atomicAdd(&esum[dIdx], expf(v - emax[dIdx]));
}

// one wave per edge; lane L handles feature cols [8L, 8L+8) (single head each)
__global__ __launch_bounds__(256)
void scatter_kernel(const int* __restrict__ ei, const float* __restrict__ el,
                    const float* __restrict__ er, const float* __restrict__ emax,
                    const float* __restrict__ esum, const float* __restrict__ feat,
                    float* __restrict__ z) {
    const int m    = blockIdx.z;
    const int wave = threadIdx.x >> 5;
    const int lane = threadIdx.x & 31;
    const int e    = blockIdx.x * 8 + wave;
    if (e >= E_EDGES) return;
    int src = ei[(size_t)m * 2 * E_EDGES + e];
    int dst = ei[(size_t)m * 2 * E_EDGES + E_EDGES + e];
    const int hh = lane >> 2;
    size_t sIdx = ((size_t)m * N_NODES + src) * H_HEADS + hh;
    size_t dIdx = ((size_t)m * N_NODES + dst) * H_HEADS + hh;
    float v = el[sIdx] + er[dIdx];
    v = v > 0.f ? v : 0.2f * v;
    float alpha = expf(v - emax[dIdx]) / esum[dIdx];
    const int cbase = lane * 8;
    const float4* f4 = (const float4*)(feat + ((size_t)m * N_NODES + src) * HD + cbase);
    float4 f0 = f4[0], f1 = f4[1];
    float* zp = z + ((size_t)dst * M_PATHS + m) * HD + cbase;   // z layout [N, M, HD]
    atomicAdd(zp + 0, alpha * f0.x); atomicAdd(zp + 1, alpha * f0.y);
    atomicAdd(zp + 2, alpha * f0.z); atomicAdd(zp + 3, alpha * f0.w);
    atomicAdd(zp + 4, alpha * f1.x); atomicAdd(zp + 5, alpha * f1.y);
    atomicAdd(zp + 6, alpha * f1.z); atomicAdd(zp + 7, alpha * f1.w);
}

__global__ void elu_bias_kernel(float* __restrict__ z, const float* __restrict__ b_gat,
                                __bf16* __restrict__ zb) {
    size_t idx = (size_t)blockIdx.x * blockDim.x + threadIdx.x;
    const size_t total = (size_t)N_NODES * M_PATHS * HD;
    if (idx >= total) return;
    int c = (int)(idx % HD);
    int m = (int)((idx / HD) % M_PATHS);
    float x = z[idx] + b_gat[m * HD + c];
    x = x > 0.f ? x : (expf(x) - 1.0f);            // elu
    z[idx]  = x;
    zb[idx] = f2bf(x);
}

// per (n,m) row: s = sum_k tanh(tmp + b1) * W2; accumulate mean numerator per m
__global__ void sem_w_kernel(const float* __restrict__ sem_tmp, const float* __restrict__ sem_b1,
                             const float* __restrict__ sem_W2, float* __restrict__ wsum) {
    int row = blockIdx.x * blockDim.x + threadIdx.x;
    if (row >= N_NODES * M_PATHS) return;
    const float* t = sem_tmp + (size_t)row * SEM;
    float s = 0.f;
    for (int k = 0; k < SEM; ++k) s += tanhf(t[k] + sem_b1[k]) * sem_W2[k];
    atomicAdd(&wsum[row % M_PATHS], s);
}

__global__ void beta_kernel(const float* __restrict__ wsum, float* __restrict__ beta) {
    if (threadIdx.x == 0 && blockIdx.x == 0) {
        float w0 = wsum[0] / (float)N_NODES;
        float w1 = wsum[1] / (float)N_NODES;
        float w2 = wsum[2] / (float)N_NODES;
        float mx = fmaxf(w0, fmaxf(w1, w2));
        float e0 = expf(w0 - mx), e1 = expf(w1 - mx), e2 = expf(w2 - mx);
        float s = e0 + e1 + e2;
        beta[0] = e0 / s; beta[1] = e1 / s; beta[2] = e2 / s;
    }
}

// one wave per node: x = sum_m beta*z ; x@lin_W(+b) ; @d1_W ; @d2_W ; sigmoid
__global__ __launch_bounds__(256)
void final_kernel(const float* __restrict__ z, const float* __restrict__ beta,
                  const float* __restrict__ lin_W, const float* __restrict__ lin_b,
                  const float* __restrict__ d1_W, const float* __restrict__ d2_W,
                  float* __restrict__ out) {
    const int wave = threadIdx.x >> 5;
    const int lane = threadIdx.x & 31;
    const int n = blockIdx.x * 8 + wave;
    if (n >= N_NODES) return;
    float b0 = beta[0], b1 = beta[1], b2 = beta[2];
    const float* zr = z + (size_t)n * M_PATHS * HD;
    float acc = 0.f;
    for (int c0 = 0; c0 < HD; c0 += 32) {
        int c = c0 + lane;
        float xv = b0 * zr[c] + b1 * zr[HD + c] + b2 * zr[2 * HD + c];
        #pragma unroll
        for (int k = 0; k < 32; ++k) {
            float xk = __shfl(xv, k, 32);          // wave32 broadcast
            acc += xk * lin_W[(size_t)(c0 + k) * OUT_F + lane];
        }
    }
    float o = acc + lin_b[lane];                   // lane j = out col j
    float t[8];
    #pragma unroll
    for (int k = 0; k < 8; ++k) t[k] = o * d1_W[lane * 8 + k];
    #pragma unroll
    for (int off = 16; off >= 1; off >>= 1) {
        #pragma unroll
        for (int k = 0; k < 8; ++k) t[k] += __shfl_xor(t[k], off, 32);
    }
    if (lane == 0) {
        float s = 0.f;
        #pragma unroll
        for (int k = 0; k < 8; ++k) s += t[k] * d2_W[k];
        out[n] = 1.0f / (1.0f + expf(-s));
    }
}

// ---------------------------------------------------------------- launch
extern "C" void kernel_launch(void* const* d_in, const int* in_sizes, int n_in,
                              void* d_out, int out_size, void* d_ws, size_t ws_size,
                              hipStream_t stream) {
    (void)in_sizes; (void)n_in; (void)out_size; (void)ws_size;
    const float* h      = (const float*)d_in[0];
    const int*   ei     = (const int*)d_in[1];
    const float* W_gat  = (const float*)d_in[2];
    const float* attn_l = (const float*)d_in[3];
    const float* attn_r = (const float*)d_in[4];
    const float* b_gat  = (const float*)d_in[5];
    const float* sem_W1 = (const float*)d_in[6];
    const float* sem_b1 = (const float*)d_in[7];
    const float* sem_W2 = (const float*)d_in[8];
    const float* lin_W  = (const float*)d_in[9];
    const float* lin_b  = (const float*)d_in[10];
    const float* d1_W   = (const float*)d_in[11];
    const float* d2_W   = (const float*)d_in[12];
    float* out = (float*)d_out;

    char* ws = (char*)d_ws;
    size_t off = 0;
    auto carve = [&](size_t bytes) -> char* {
        char* p = ws + off;
        off = (off + bytes + 255) & ~(size_t)255;
        return p;
    };
    __bf16* h_bf     = (__bf16*)carve((size_t)N_NODES * IN_F * 2);
    __bf16* WT_bf    = (__bf16*)carve((size_t)M_PATHS * HD * IN_F * 2);
    __bf16* semW1T   = (__bf16*)carve((size_t)SEM * HD * 2);
    float*  feat     = (float*) carve((size_t)M_PATHS * N_NODES * HD * 4);
    float*  el       = (float*) carve((size_t)M_PATHS * N_NODES * H_HEADS * 4);
    float*  er       = (float*) carve((size_t)M_PATHS * N_NODES * H_HEADS * 4);
    float*  emax     = (float*) carve((size_t)M_PATHS * N_NODES * H_HEADS * 4);
    float*  esum     = (float*) carve((size_t)M_PATHS * N_NODES * H_HEADS * 4);
    float*  zbuf     = (float*) carve((size_t)N_NODES * M_PATHS * HD * 4);
    __bf16* z_bf     = (__bf16*)carve((size_t)N_NODES * M_PATHS * HD * 2);
    float*  sem_tmp  = (float*) carve((size_t)N_NODES * M_PATHS * SEM * 4);
    float*  wsum     = (float*) carve(16);
    float*  betab    = (float*) carve(16);

    const size_t nH   = (size_t)N_NODES * IN_F;
    const size_t nNH  = (size_t)M_PATHS * N_NODES * H_HEADS;
    const size_t nZ   = (size_t)N_NODES * M_PATHS * HD;

    // converts (bf16 for WMMA; weights pre-transposed so B fragments load contiguously)
    cvt_bf16_kernel<<<dim3((unsigned)((nH + 255) / 256)), 256, 0, stream>>>(h, h_bf, nH);
    cvt_bf16_T_kernel<<<dim3((IN_F * HD + 255) / 256, 1, M_PATHS), 256, 0, stream>>>(W_gat, WT_bf, IN_F, HD);
    cvt_bf16_T_kernel<<<dim3((HD * SEM + 255) / 256, 1, 1), 256, 0, stream>>>(sem_W1, semW1T, HD, SEM);

    // init accumulators
    fill_u32_kernel<<<dim3((unsigned)((nNH + 255) / 256)), 256, 0, stream>>>((unsigned*)emax, 0xFF800000u, nNH);
    fill_u32_kernel<<<dim3((unsigned)((nNH + 255) / 256)), 256, 0, stream>>>((unsigned*)esum, 0u, nNH);
    fill_u32_kernel<<<dim3((unsigned)((nZ + 255) / 256)), 256, 0, stream>>>((unsigned*)zbuf, 0u, nZ);
    fill_u32_kernel<<<1, 256, 0, stream>>>((unsigned*)wsum, 0u, 4);

    // feat[m] = h @ W[m]  (WMMA bf16 -> f32), 32x32 macro tiles per wave
    {
        int macroTiles = (N_NODES + 31) / 32;               // 1563 (last is 16 rows)
        dim3 grid((macroTiles + 7) / 8, HD / 32, M_PATHS);
        gemm_bf16_wmma_kernel<<<grid, 256, 0, stream>>>(
            h_bf, WT_bf, feat, N_NODES, IN_F, HD,
            (long long)HD * IN_F, (long long)N_NODES * HD);
    }

    el_er_kernel<<<dim3((unsigned)((nNH + 255) / 256)), 256, 0, stream>>>(feat, attn_l, attn_r, el, er);

    {
        dim3 grid((E_EDGES * H_HEADS + 255) / 256, 1, M_PATHS);
        edge_max_kernel<<<grid, 256, 0, stream>>>(ei, el, er, emax);
        edge_sum_kernel<<<grid, 256, 0, stream>>>(ei, el, er, emax, esum);
    }

    scatter_kernel<<<dim3(E_EDGES / 8, 1, M_PATHS), 256, 0, stream>>>(ei, el, er, emax, esum, feat, zbuf);

    elu_bias_kernel<<<dim3((unsigned)((nZ + 255) / 256)), 256, 0, stream>>>(zbuf, b_gat, z_bf);

    // sem_tmp = z @ sem_W1  (WMMA bf16 -> f32), rows = N*M = 150000
    {
        int R = N_NODES * M_PATHS;
        int macroTiles = (R + 31) / 32;                     // 4688 (last is 16 rows)
        dim3 grid((macroTiles + 7) / 8, SEM / 32, 1);
        gemm_bf16_wmma_kernel<<<grid, 256, 0, stream>>>(z_bf, semW1T, sem_tmp, R, HD, SEM, 0, 0);
    }

    sem_w_kernel<<<dim3((N_NODES * M_PATHS + 255) / 256), 256, 0, stream>>>(sem_tmp, sem_b1, sem_W2, wsum);
    beta_kernel<<<1, 32, 0, stream>>>(wsum, betab);

    final_kernel<<<dim3(N_NODES / 8), 256, 0, stream>>>(zbuf, betab, lin_W, lin_b, d1_W, d2_W, out);
}
